// PerformerBlock_71528385348141
// MI455X (gfx1250) — matrix operand order repaired and checked by reading, not compiled, not run on hardware
//
#include <hip/hip_runtime.h>
#include <hip/hip_bf16.h>

typedef __attribute__((ext_vector_type(16))) _Float16 v16h;
typedef __attribute__((ext_vector_type(8)))  float    v8f;
typedef __attribute__((ext_vector_type(4)))  float    f4;
typedef __attribute__((ext_vector_type(4)))  _Float16 h4;

// Problem constants (from reference)
#define BB   2
#define NN   4096
#define DIM  1024
#define HH   16
#define DHH  64
#define FF   256

// Epilogue selectors
#define EPI_NONE     0
#define EPI_BIAS     1
#define EPI_EXP      2
#define EPI_ROWSCALE 3

// ---------------------------------------------------------------------------
// Generic batched strided GEMM:  C[z] = epi( A[z] (MxK) * B[z] (KxN) )
//   A(m,k) at A + zb*aB0 + zh*aB1 + m*sAm + k*sAk    (zb=z/HB, zh=z%HB)
//   B(k,n) at B + zb*bB0 + zh*bB1 + k*sBk + n       (B n-contiguous)
//   C(m,n) at C + zb*cB0 + zh*cB1 + m*ldc + n
// WN = waves along N: 4 -> 128x128 block tile, 2 -> 256x64 block tile.
// Requirements: M % BM == 0, K % 32 == 0; N handled branchless per 4-col group.
// AMODE 0: sAk==1 (f4 loads along k).  AMODE 1: sAm==1 (f4 loads along m).
// Software-pipelined: tile i+1 prefetched to registers during tile i's WMMAs.
// ---------------------------------------------------------------------------
template<int EPI, int AMODE, int WN>
__global__ __launch_bounds__(256)
void gemm_wmma_kernel(const float* __restrict__ A, const float* __restrict__ B,
                      const float* __restrict__ bias,
                      const float* __restrict__ rowscale,
                      float* __restrict__ C,
                      int M, int N, int K,
                      long long sAm, long long sAk, long long aB0, long long aB1,
                      long long sBk, long long bB0, long long bB1,
                      long long ldc, long long cB0, long long cB1,
                      long long rsB0, long long rsB1, int HB)
{
    constexpr int WM   = 8 / WN;        // waves along M
    constexpr int BM   = WM * 64;       // block rows
    constexpr int BN   = WN * 32;       // block cols
    constexpr int A_IT = BM / 32;       // f4 staging iterations for A
    constexpr int B_IT = BN / 32;       // f4 staging iterations for B

    __shared__ _Float16 As[BM][40];     // [m][k], +8 half pad
    __shared__ _Float16 Bs[BN][40];     // [n][k] (transposed), +8 half pad

    const int tid    = threadIdx.x;
    const int lane   = tid & 31;
    const int wv     = tid >> 5;
    const int wave_m = wv / WN;         // 0..WM-1 -> 64-row slab
    const int wave_n = wv % WN;         // 0..WN-1 -> 32-col slab
    const int hi     = lane >> 4;       // lane half (0 or 1)
    const int ln     = lane & 15;

    const int z  = blockIdx.z;
    const int zb = z / HB, zh = z % HB;
    const float* Ab = A + (long long)zb * aB0 + (long long)zh * aB1;
    const float* Bb = B + (long long)zb * bB0 + (long long)zh * bB1;
    float*       Cb = C + (long long)zb * cB0 + (long long)zh * cB1;
    const long long rsb = (long long)zb * rsB0 + (long long)zh * rsB1;

    const long long bm0 = (long long)blockIdx.y * BM;
    const long long bn0 = (long long)blockIdx.x * BN;

    // ---- per-thread staging pointers (hoisted out of the K loop) ----------
    const float* aptr[A_IT];
    int a_lm[A_IT], a_lk[A_IT];
    const long long aStep = (AMODE == 0) ? 32 : 32 * sAk;
#pragma unroll
    for (int i = 0; i < A_IT; ++i) {
        int idx = i * 256 + tid;
        if (AMODE == 0) {               // vector along k: tile [m][k4]
            int m = idx >> 3, k4 = idx & 7;
            a_lm[i] = m; a_lk[i] = 4 * k4;
            aptr[i] = Ab + (bm0 + m) * sAm + 4 * k4;
        } else {                        // vector along m: tile [m4][k]
            int m4 = idx & (BM / 4 - 1), kk = idx / (BM / 4);
            a_lm[i] = 4 * m4; a_lk[i] = kk;
            aptr[i] = Ab + (bm0 + 4 * m4) + (long long)kk * sAk;
        }
    }
    const float* bptr[B_IT];
    int b_ln4[B_IT], b_lk[B_IT];
    bool b_ok[B_IT];
#pragma unroll
    for (int i = 0; i < B_IT; ++i) {
        int idx = i * 256 + tid;
        int n4 = idx & (BN / 4 - 1), kk = idx / (BN / 4);
        long long gn = bn0 + 4 * n4;
        b_ok[i]  = gn < N;              // N is a multiple of 4 in all launches
        b_ln4[i] = 4 * n4; b_lk[i] = kk;
        bptr[i]  = Bb + (long long)kk * sBk + (b_ok[i] ? gn : 0);
    }
    const long long bStep = 32 * sBk;

    v8f acc[4][2];
#pragma unroll
    for (int i = 0; i < 4; ++i)
#pragma unroll
        for (int j = 0; j < 2; ++j) acc[i][j] = (v8f){0,0,0,0,0,0,0,0};

    // ---- prologue: prefetch tile 0 into registers -------------------------
    f4 aReg[A_IT], bReg[B_IT];
#pragma unroll
    for (int i = 0; i < A_IT; ++i) { aReg[i] = *(const f4*)aptr[i]; aptr[i] += aStep; }
#pragma unroll
    for (int i = 0; i < B_IT; ++i) { bReg[i] = *(const f4*)bptr[i]; bptr[i] += bStep; }

    for (int k0 = 0; k0 < K; k0 += 32) {
        // ---- commit prefetched tile to LDS (f32 -> f16) ------------------
#pragma unroll
        for (int i = 0; i < A_IT; ++i) {
            f4 v = aReg[i];
            if (AMODE == 0) {
                h4 hv = { (_Float16)v.x, (_Float16)v.y,
                          (_Float16)v.z, (_Float16)v.w };
                *(h4*)&As[a_lm[i]][a_lk[i]] = hv;        // ds_store_b64
            } else {
                As[a_lm[i] + 0][a_lk[i]] = (_Float16)v.x;
                As[a_lm[i] + 1][a_lk[i]] = (_Float16)v.y;
                As[a_lm[i] + 2][a_lk[i]] = (_Float16)v.z;
                As[a_lm[i] + 3][a_lk[i]] = (_Float16)v.w;
            }
        }
#pragma unroll
        for (int i = 0; i < B_IT; ++i) {
            f4 v = bReg[i];
            if (!b_ok[i]) v = (f4){0.f, 0.f, 0.f, 0.f};  // branchless mask
            Bs[b_ln4[i] + 0][b_lk[i]] = (_Float16)v.x;
            Bs[b_ln4[i] + 1][b_lk[i]] = (_Float16)v.y;
            Bs[b_ln4[i] + 2][b_lk[i]] = (_Float16)v.z;
            Bs[b_ln4[i] + 3][b_lk[i]] = (_Float16)v.w;
        }
        __syncthreads();

        // ---- prefetch next tile (overlaps with WMMA below) ---------------
        if (k0 + 32 < K) {
#pragma unroll
            for (int i = 0; i < A_IT; ++i) { aReg[i] = *(const f4*)aptr[i]; aptr[i] += aStep; }
#pragma unroll
            for (int i = 0; i < B_IT; ++i) { bReg[i] = *(const f4*)bptr[i]; bptr[i] += bStep; }
        }

        // ---- build fragments (ISA wave32 16-bit layouts) and multiply ----
        v16h af[4];
#pragma unroll
        for (int tm = 0; tm < 4; ++tm) {
            int mrow = wave_m * 64 + tm * 16 + ln;
#pragma unroll
            for (int i = 0; i < 8; ++i) {
                int k = ((i < 4) ? 0 : 16) + hi * 8 + 2 * (i & 3);
                af[tm][2 * i]     = As[mrow][k];
                af[tm][2 * i + 1] = As[mrow][k + 1];
            }
        }
        v16h bf[2];
#pragma unroll
        for (int tn = 0; tn < 2; ++tn) {
            int ncol = wave_n * 32 + tn * 16 + ln;
            int kb = hi * 8;
#pragma unroll
            for (int i = 0; i < 16; ++i) bf[tn][i] = Bs[ncol][kb + (i & 7) + ((i < 8) ? 0 : 16)];
        }
#pragma unroll
        for (int tm = 0; tm < 4; ++tm)
#pragma unroll
            for (int tn = 0; tn < 2; ++tn)
                acc[tm][tn] = __builtin_amdgcn_wmma_f32_16x16x32_f16(
                    false, af[tm], false, bf[tn], (short)0, acc[tm][tn],
                    false, false);
        __syncthreads();
    }

    // ---- epilogue + store (C/D: lanes 0-15 -> M=r, lanes 16-31 -> M=r+8)
#pragma unroll
    for (int tm = 0; tm < 4; ++tm)
#pragma unroll
        for (int tn = 0; tn < 2; ++tn) {
            long long gcol = bn0 + wave_n * 32 + tn * 16 + ln;
            if (gcol < N) {
#pragma unroll
                for (int r = 0; r < 8; ++r) {
                    long long grow = bm0 + wave_m * 64 + tm * 16 + hi * 8 + r;
                    float val = acc[tm][tn][r];
                    if constexpr (EPI == EPI_BIAS)     val += bias[gcol];
                    if constexpr (EPI == EPI_EXP)      val = __expf(val);
                    if constexpr (EPI == EPI_ROWSCALE) val *= rowscale[rsb + grow];
                    Cb[grow * ldc + gcol] = val;
                }
            }
        }
}

// k_sum[b,h,f] = sum_n k_prime[b,n,h,f]      (one thread per (b,h,f))
__global__ void ksum_kernel(const float* __restrict__ kp, float* __restrict__ ksum)
{
    int t = blockIdx.x * blockDim.x + threadIdx.x;
    if (t >= BB * HH * FF) return;
    int f = t % FF, h = (t / FF) % HH, b = t / (FF * HH);
    const float* p = kp + (((long long)b * NN) * HH + h) * FF + f;
    const long long stride = (long long)HH * FF;
    float acc = 0.f;
    for (int n = 0; n < NN; ++n) acc += p[(long long)n * stride];
    ksum[t] = acc;                       // [B,H,F]
}

// dinv[b,h,n] = 1/(dot(q_prime[b,n,h,:], ksum[b,h,:]) + 1e-6)  (one wave32 per row)
__global__ void dinv_kernel(const float* __restrict__ qp, const float* __restrict__ ksum,
                            float* __restrict__ dinv)
{
    int lane = threadIdx.x & 31;
    int warp = threadIdx.x >> 5;
    long long w = (long long)blockIdx.x * (blockDim.x >> 5) + warp;
    if (w >= (long long)BB * HH * NN) return;
    int n = (int)(w % NN);
    int h = (int)((w / NN) % HH);
    int b = (int)(w / ((long long)NN * HH));
    const float* q = qp + (((long long)b * NN + n) * HH + h) * FF;
    const float* s = ksum + ((long long)b * HH + h) * FF;
    float acc = 0.f;
    for (int f = lane; f < FF; f += 32) acc += q[f] * s[f];
    for (int o = 16; o > 0; o >>= 1) acc += __shfl_xor(acc, o, 32);
    if (lane == 0) dinv[w] = 1.0f / (acc + 1e-6f);   // [B,H,N]
}

extern "C" void kernel_launch(void* const* d_in, const int* in_sizes, int n_in,
                              void* d_out, int out_size, void* d_ws, size_t ws_size,
                              hipStream_t stream)
{
    const float* x      = (const float*)d_in[0];
    const float* Wq     = (const float*)d_in[1];
    const float* bq     = (const float*)d_in[2];
    const float* Wk     = (const float*)d_in[3];
    const float* bk     = (const float*)d_in[4];
    const float* Wv     = (const float*)d_in[5];
    const float* bv     = (const float*)d_in[6];
    const float* Wproj  = (const float*)d_in[7];
    const float* bproj  = (const float*)d_in[8];
    const float* wqf    = (const float*)d_in[9];
    const float* wkf    = (const float*)d_in[10];
    float* out = (float*)d_out;

    const long long M1 = (long long)BB * NN;        // 8192
    const long long MR = (long long)BB * NN * HH;   // 131072 feature rows

    float* ws = (float*)d_ws;
    size_t o = 0;
    float* q    = ws + o; o += (size_t)M1 * (HH * DHH);
    float* k    = ws + o; o += (size_t)M1 * (HH * DHH);
    float* v    = ws + o; o += (size_t)M1 * (HH * DHH);
    float* qp   = ws + o; o += (size_t)MR * FF;
    float* kp   = ws + o; o += (size_t)MR * FF;
    float* ksum = ws + o; o += (size_t)BB * HH * FF;
    float* dinv = ws + o; o += (size_t)MR;
    float* ctx  = ws + o; o += (size_t)BB * HH * FF * DHH;
    float* attn = q;   // q is dead after q_prime is formed

    dim3 blk(256);

    // ---- QKV projections: [8192,1024] x [1024,1024] + bias  (128x128 tiles)
    for (int which = 0; which < 3; ++which) {
        const float* W = which == 0 ? Wq : which == 1 ? Wk : Wv;
        const float* bvec = which == 0 ? bq : which == 1 ? bk : bv;
        float* dst = which == 0 ? q : which == 1 ? k : v;
        gemm_wmma_kernel<EPI_BIAS, 0, 4><<<dim3(8, 64, 1), blk, 0, stream>>>(
            x, W, bvec, nullptr, dst,
            (int)M1, HH * DHH, DIM,
            DIM, 1, 0, 0,
            HH * DHH, 0, 0,
            HH * DHH, 0, 0, 0, 0, 1);
    }

    // ---- feature maps: exp([131072,64] x [64,256])  (128x128 tiles)
    gemm_wmma_kernel<EPI_EXP, 0, 4><<<dim3(2, 1024, 1), blk, 0, stream>>>(
        q, wqf, nullptr, nullptr, qp,
        (int)MR, FF, DHH,
        DHH, 1, 0, 0,
        FF, 0, 0,
        FF, 0, 0, 0, 0, 1);
    gemm_wmma_kernel<EPI_EXP, 0, 4><<<dim3(2, 1024, 1), blk, 0, stream>>>(
        k, wkf, nullptr, nullptr, kp,
        (int)MR, FF, DHH,
        DHH, 1, 0, 0,
        FF, 0, 0,
        FF, 0, 0, 0, 0, 1);

    // ---- k_sum [B,H,F] and D_inv [B,H,N]
    ksum_kernel<<<dim3((BB * HH * FF + 255) / 256), blk, 0, stream>>>(kp, ksum);
    dinv_kernel<<<dim3((unsigned)((MR + 7) / 8)), blk, 0, stream>>>(qp, ksum, dinv);

    // ---- context[b,h] = k_prime^T (F x N) * v (N x DH), batch B*H  (256x64)
    gemm_wmma_kernel<EPI_NONE, 1, 2><<<dim3(1, 1, BB * HH), blk, 0, stream>>>(
        kp, v, nullptr, nullptr, ctx,
        FF, DHH, NN,
        1, (long long)HH * FF, (long long)NN * HH * FF, FF,
        (long long)HH * DHH, (long long)NN * HH * DHH, DHH,
        DHH, (long long)HH * FF * DHH, (long long)FF * DHH,
        0, 0, HH);

    // ---- attn[b,h] = D_inv * (q_prime (N x F) * context (F x DH))  (256x64)
    gemm_wmma_kernel<EPI_ROWSCALE, 0, 2><<<dim3(1, 16, BB * HH), blk, 0, stream>>>(
        qp, ctx, nullptr, dinv, attn,
        NN, DHH, FF,
        (long long)HH * FF, 1, (long long)NN * HH * FF, FF,
        DHH, (long long)HH * FF * DHH, (long long)FF * DHH,
        (long long)HH * DHH, (long long)NN * HH * DHH, DHH,
        (long long)HH * NN, NN, HH);

    // ---- output projection: [8192,1024] x [1024,1024] + bias -> d_out
    gemm_wmma_kernel<EPI_BIAS, 0, 4><<<dim3(8, 64, 1), blk, 0, stream>>>(
        attn, Wproj, bproj, nullptr, out,
        (int)M1, DIM, HH * DHH,
        HH * DHH, 1, 0, 0,
        DIM, 0, 0,
        DIM, 0, 0, 0, 0, 1);
}
